// Snook_50775103373857
// MI455X (gfx1250) — matrix operand order repaired and checked by reading, not compiled, not run on hardware
//
#include <hip/hip_runtime.h>
#include <math.h>

#define HH 4096
#define WW 4096
#define MAXP 8192
#define ALPHA 0.9998f
#define LALPHA 0.3f
#define KTOT 3072
#define NCHUNK 96
#define NSEG 4
#define CPS 24          // chunks per K-segment (96/4)
#define NTILE 512       // 8192/16

#define OUT_SCORES 40960
#define OUT_CUE    40976
#define OUT_PEAKS  40980

typedef __attribute__((ext_vector_type(16))) _Float16 v16h;
typedef __attribute__((ext_vector_type(8)))  float    v8f;
// 4-byte-aligned float4 so b128 loads are emitted without claiming 16B alignment
typedef float f32x4 __attribute__((ext_vector_type(4), aligned(4)));

union V16H { v16h v; _Float16 e[16]; };
union V8F  { v8f  v; float    e[8];  };
union F4   { f32x4 v; float   e[4];  };

// ---- clamped heatmap value with zero padding outside [0,H)x[0,W) ----
__device__ __forceinline__ float hval(const float* hm, int y, int x) {
    if ((unsigned)y >= HH || (unsigned)x >= WW) return 0.0f;
    float v = hm[(size_t)y * WW + x];
    v = fminf(fmaxf(v, 0.0f), 1.0f);
    return (v < ALPHA) ? 0.0f : v;
}

__device__ __forceinline__ bool is_peak(const float* hm, int y, int x) {
    float c = hval(hm, y, x);
    return (c > hval(hm, y, x + 1)) && (c > hval(hm, y, x - 1)) &&
           (c > hval(hm, y + 1, x)) && (c > hval(hm, y - 1, x));
}

// ---- 1: fill coords with -1 ----
__global__ void init_kernel(int* coords) {
    int i = blockIdx.x * blockDim.x + threadIdx.x;
    if (i < 2 * MAXP) coords[i] = -1;
}

// ---- 2: per-row peak counts (one wave32 per row) ----
__global__ void rowcount_kernel(const float* __restrict__ hm, int* __restrict__ rowcnt) {
    int row  = (blockIdx.x * blockDim.x + threadIdx.x) >> 5;
    int lane = threadIdx.x & 31;
    if (row >= HH) return;
    int cnt = 0;
    for (int xb = 0; xb < WW; xb += 32) {
        bool p = is_peak(hm, row, xb + lane);
        unsigned long long b = __ballot(p);
        cnt += __popcll(b);
    }
    if (lane == 0) rowcnt[row] = cnt;
}

// ---- 3: exclusive prefix scan of 4096 row counts (single block) ----
__global__ void scan_kernel(const int* __restrict__ rc, int* __restrict__ ro) {
    __shared__ int part[1024];
    int t = threadIdx.x;
    int v0 = rc[4 * t], v1 = rc[4 * t + 1], v2 = rc[4 * t + 2], v3 = rc[4 * t + 3];
    int s = v0 + v1 + v2 + v3;
    part[t] = s;
    __syncthreads();
    for (int off = 1; off < 1024; off <<= 1) {
        int vv = (t >= off) ? part[t - off] : 0;
        __syncthreads();
        part[t] += vv;
        __syncthreads();
    }
    int excl = part[t] - s;
    ro[4 * t]     = excl;
    ro[4 * t + 1] = excl + v0;
    ro[4 * t + 2] = excl + v0 + v1;
    ro[4 * t + 3] = excl + v0 + v1 + v2;
}

// ---- 4: compact peaks in row-major order (one wave per row) ----
__global__ void compact_kernel(const float* __restrict__ hm, const int* __restrict__ rowoff,
                               int* __restrict__ coords) {
    int row  = (blockIdx.x * blockDim.x + threadIdx.x) >> 5;
    int lane = threadIdx.x & 31;
    if (row >= HH) return;
    int base = rowoff[row];
    int run  = 0;
    unsigned long long ltmask = (1ull << lane) - 1ull;
    for (int xb = 0; xb < WW; xb += 32) {
        int x = xb + lane;
        bool p = is_peak(hm, row, x);
        unsigned long long b = __ballot(p);
        if (p) {
            int r = base + run + __popcll(b & ltmask);
            if (r < MAXP) { coords[2 * r] = row; coords[2 * r + 1] = x; }
        }
        run += __popcll(b);
    }
}

// ---- 5: NMS (exact reference formula) ----
__global__ void nms_kernel(const int* __restrict__ coords, int* __restrict__ keep) {
    int i  = blockIdx.x * blockDim.x + threadIdx.x;
    int yi = coords[2 * i], xi = coords[2 * i + 1];
    float fy = (float)yi, fx = (float)xi;
    float n2i = fy * fy + fx * fx;
    __shared__ float sy[256], sx[256], sn[256];
    __shared__ int   sv[256];
    bool rej = false;
    for (int tb = 0; tb < MAXP; tb += 256) {
        int j  = tb + threadIdx.x;
        int yj = coords[2 * j], xj = coords[2 * j + 1];
        __syncthreads();
        float fyj = (float)yj, fxj = (float)xj;
        sy[threadIdx.x] = fyj; sx[threadIdx.x] = fxj;
        sn[threadIdx.x] = fyj * fyj + fxj * fxj;
        sv[threadIdx.x] = (yj >= 0);
        __syncthreads();
        for (int u = 0; u < 256; ++u) {
            float d2 = n2i + sn[u] - 2.0f * (fy * sy[u] + fx * sx[u]);
            rej = rej || ((d2 > 1.0f) && (d2 < 25.0f) && sv[u]);
        }
    }
    keep[i] = (yi >= 0) && !rej;
}

// ---- 6: pack [label_W | dir_W | 0-pad] into f16 WMMA B-fragment layout ----
// B is 32x16 (KxN) per chunk. lanes 0-15: N=lane, halves j -> K=chunk*32+(j<8 ? j : 16+j-8);
// lanes 16-31: N=lane-16, halves j -> K=chunk*32+(j<8 ? 8+j : 24+j-8).
__global__ void bpack_kernel(const float* __restrict__ lw, const float* __restrict__ dw,
                             _Float16* __restrict__ bp) {
    int t = blockIdx.x * blockDim.x + threadIdx.x;   // 0..3071 == chunk*32+lane
    if (t >= NCHUNK * 32) return;
    int ch = t >> 5, lane = t & 31;
    int n = lane & 15;
    int koff = (lane < 16) ? 0 : 8;
    for (int j = 0; j < 16; ++j) {
        int K = ch * 32 + koff + ((j < 8) ? j : j + 8);
        float v = 0.0f;
        if (n < 5)      v = lw[K * 5 + n];
        else if (n < 7) v = dw[K * 2 + (n - 5)];
        bp[(size_t)t * 16 + j] = (_Float16)v;
    }
}

// ---- 7: window-gather GEMM via v_wmma_f32_16x16x32_f16 ----
// wave -> (tile of 16 peaks, K-segment). Each 32-wide K chunk == one window row,
// i.e. two contiguous 8-float runs per lane. Common case (peak >=16px from the
// border): four global_load_b128 per chunk. Rare border lanes fall back to
// clamped scalar loads + zero-selects (still branch-free per element).
__global__ void gemm_kernel(const float* __restrict__ img, const int* __restrict__ coords,
                            const _Float16* __restrict__ bpack, float* __restrict__ partial) {
    int gid  = blockIdx.x * blockDim.x + threadIdx.x;
    int wave = gid >> 5;
    int lane = threadIdx.x & 31;
    int tile = wave >> 2;
    int seg  = wave & 3;
    if (tile >= NTILE) return;
    bool hi = (lane >= 16);
    int m    = lane & 15;
    int peak = tile * 16 + m;
    int py = coords[2 * peak];     if (py < 0) py = 0;
    int px = coords[2 * peak + 1]; if (px < 0) px = 0;
    int y0 = py - 16, x0 = px - 16;
    int xa = x0 + (hi ? 8 : 0);       // halves 0..7  -> x = xa+j
                                      // halves 8..15 -> x = xa+16+(j-8)
    bool xok = (xa >= 0) && (xa + 23 < WW);   // both 8-float runs fully in-bounds

    V8F acc;
#pragma unroll
    for (int r = 0; r < 8; ++r) acc.e[r] = 0.0f;

    int c0 = seg * CPS;
#pragma unroll 2
    for (int ch = c0; ch < c0 + CPS; ++ch) {
        int kk = ch << 5;
        int cc = kk >> 10;
        int wy = (kk >> 5) & 31;
        int y  = y0 + wy;
        int yc = y; if (yc < 0) yc = 0; if (yc > HH - 1) yc = HH - 1;
        bool yok = ((unsigned)y < HH);
        const float* row = img + ((size_t)cc * HH + (size_t)yc) * WW;

        // branch-free prefetch of next chunk's window row
        {
            int ch2 = (ch + 1 < c0 + CPS) ? ch + 1 : ch;
            int kk2 = ch2 << 5;
            int y2  = y0 + ((kk2 >> 5) & 31);
            if (y2 < 0) y2 = 0; if (y2 > HH - 1) y2 = HH - 1;
            int cc2 = kk2 >> 10;
            int xp  = (x0 < 0) ? 0 : x0;
            __builtin_prefetch(img + ((size_t)cc2 * HH + (size_t)y2) * WW + xp, 0, 3);
        }

        V16H a;
        if (xok && yok) {
            // fast path: 4 x b128 (4-byte-aligned vector loads)
            F4 r0, r1, r2, r3;
            r0.v = *(const f32x4*)(row + xa);
            r1.v = *(const f32x4*)(row + xa + 4);
            r2.v = *(const f32x4*)(row + xa + 16);
            r3.v = *(const f32x4*)(row + xa + 20);
#pragma unroll
            for (int j = 0; j < 4; ++j) {
                a.e[j]      = (_Float16)r0.e[j];
                a.e[4 + j]  = (_Float16)r1.e[j];
                a.e[8 + j]  = (_Float16)r2.e[j];
                a.e[12 + j] = (_Float16)r3.e[j];
            }
        } else {
            // slow path: clamped scalar loads, zero OOB with selects
            float raw[16];
#pragma unroll
            for (int j = 0; j < 16; ++j) {
                int x  = xa + ((j < 8) ? j : j + 8);
                int xc = x; if (xc < 0) xc = 0; if (xc > WW - 1) xc = WW - 1;
                raw[j] = row[xc];
            }
#pragma unroll
            for (int j = 0; j < 16; ++j) {
                int x = xa + ((j < 8) ? j : j + 8);
                bool ok = yok && ((unsigned)x < WW);
                a.e[j] = (_Float16)(ok ? raw[j] : 0.0f);
            }
        }

        v16h b = *(const v16h*)(bpack + (size_t)(ch * 32 + lane) * 16);
        acc.v = __builtin_amdgcn_wmma_f32_16x16x32_f16(
            false, a.v, false, b, (short)0, acc.v, false, false);
    }
    // D layout: VGPR r -> lanes 0-15: (M=r, N=lane); lanes 16-31: (M=r+8, N=lane-16)
    float* dst = partial + (size_t)(seg * MAXP + tile * 16) * 16;
#pragma unroll
    for (int r = 0; r < 8; ++r) {
        int mm = r + (hi ? 8 : 0);
        dst[mm * 16 + (lane & 15)] = acc.e[r];
    }
}

// ---- 8: deterministic partial-sum + softmax head ----
__global__ void head_kernel(const float* __restrict__ partial, const int* __restrict__ keep,
                            const float* __restrict__ label_b, const float* __restrict__ dir_b,
                            float* __restrict__ out, float* __restrict__ score,
                            int* __restrict__ label, int* __restrict__ validf,
                            float* __restrict__ dirv) {
    int i = blockIdx.x * blockDim.x + threadIdx.x;
    if (i >= MAXP) return;
    float l[7];
    for (int n = 0; n < 7; ++n) {
        float s = 0.0f;
        for (int seg = 0; seg < NSEG; ++seg)
            s += partial[(size_t)(seg * MAXP + i) * 16 + n];
        l[n] = s;
    }
    float lab[5], mx = -__builtin_inff();
    for (int n = 0; n < 5; ++n) { lab[n] = l[n] + label_b[n]; mx = fmaxf(mx, lab[n]); }
    float se = 0.0f;
    for (int n = 0; n < 5; ++n) se += __expf(lab[n] - mx);
    float inv = 1.0f / se;
    float best = -1.0f; int bl = 0;
    float p[5];
    for (int n = 0; n < 5; ++n) {
        p[n] = __expf(lab[n] - mx) * inv;
        if (p[n] > best) { best = p[n]; bl = n; }
    }
    int v = keep[i] && (best > LALPHA);
    for (int n = 0; n < 5; ++n) out[(size_t)i * 5 + n] = v ? p[n] : 0.0f;
    score[i]  = best;
    label[i]  = bl;
    validf[i] = v;
    dirv[2 * i]     = l[5] + dir_b[0];
    dirv[2 * i + 1] = l[6] + dir_b[1];
}

// ---- 9: per-class top-k (masked argmax, tie-break lowest index) + outputs ----
__global__ void topk_kernel(const float* __restrict__ score, const int* __restrict__ label,
                            const int* __restrict__ validf, const float* __restrict__ dirv,
                            const int* __restrict__ coords, int* __restrict__ taken,
                            float* __restrict__ out) {
    __shared__ float sval[1024];
    __shared__ int   sidx[1024];
    __shared__ int   selIdx[16];
    int t = threadIdx.x;
    for (int i = t; i < MAXP; i += 1024) taken[i] = 0;
    __syncthreads();
    const int KC[5] = {1, 1, 6, 6, 2};
    int slot = 0;
    for (int c = 0; c < 5; ++c) {
        for (int k = 0; k < KC[c]; ++k) {
            float best = -__builtin_inff();
            int bidx = 0x7fffffff;
            for (int i = t; i < MAXP; i += 1024) {
                float s = (validf[i] && (label[i] == c) && !taken[i]) ? score[i]
                                                                     : -__builtin_inff();
                if (s > best || (s == best && i < bidx)) { best = s; bidx = i; }
            }
            sval[t] = best; sidx[t] = bidx;
            __syncthreads();
            for (int off = 512; off > 0; off >>= 1) {
                if (t < off) {
                    float ov = sval[t + off]; int oi = sidx[t + off];
                    if (ov > sval[t] || (ov == sval[t] && oi < sidx[t])) {
                        sval[t] = ov; sidx[t] = oi;
                    }
                }
                __syncthreads();
            }
            if (t == 0) {
                int idx = sidx[0]; float v = sval[0];
                selIdx[slot] = idx;
                out[OUT_SCORES + slot] = (v > -3.0e38f && v < 3.0e38f) ? v : 0.0f;
                taken[idx] = 1;
            }
            __syncthreads();
            slot++;
        }
    }
    if (t == 0) {
        int i14 = selIdx[14], i15 = selIdx[15];   // class-4 top-2; flip rows
        out[OUT_CUE + 0] = dirv[2 * i15 + 0];
        out[OUT_CUE + 1] = dirv[2 * i15 + 1];
        out[OUT_CUE + 2] = dirv[2 * i14 + 0];
        out[OUT_CUE + 3] = dirv[2 * i14 + 1];
    }
    if (t < 16) {
        int idx = selIdx[t];
        out[OUT_PEAKS + 2 * t]     = (float)coords[2 * idx];
        out[OUT_PEAKS + 2 * t + 1] = (float)coords[2 * idx + 1];
    }
}

extern "C" void kernel_launch(void* const* d_in, const int* in_sizes, int n_in,
                              void* d_out, int out_size, void* d_ws, size_t ws_size,
                              hipStream_t stream) {
    const float* hm      = (const float*)d_in[0];
    const float* image   = (const float*)d_in[1];
    const float* label_W = (const float*)d_in[2];
    const float* label_b = (const float*)d_in[3];
    const float* dir_W   = (const float*)d_in[4];
    const float* dir_b   = (const float*)d_in[5];
    float* out = (float*)d_out;

    // workspace carve-up (all region sizes multiples of 256 B; d_ws 256-B aligned)
    int* coords = (int*)d_ws;                 // 16384 ints
    int* rowcnt = coords + 2 * MAXP;          // 4096
    int* rowoff = rowcnt + HH;                // 4096
    int* keep   = rowoff + HH;                // 8192
    int* label  = keep + MAXP;                // 8192
    int* validf = label + MAXP;               // 8192
    int* taken  = validf + MAXP;              // 8192
    float* score   = (float*)(taken + MAXP);  // 8192
    float* dirv    = score + MAXP;            // 16384
    float* partial = dirv + 2 * MAXP;         // 4*8192*16 floats
    _Float16* bpack = (_Float16*)(partial + (size_t)NSEG * MAXP * 16); // 96*32*16 halves

    init_kernel<<<64, 256, 0, stream>>>(coords);
    rowcount_kernel<<<512, 256, 0, stream>>>(hm, rowcnt);
    scan_kernel<<<1, 1024, 0, stream>>>(rowcnt, rowoff);
    compact_kernel<<<512, 256, 0, stream>>>(hm, rowoff, coords);
    nms_kernel<<<32, 256, 0, stream>>>(coords, keep);
    bpack_kernel<<<12, 256, 0, stream>>>(label_W, dir_W, bpack);
    gemm_kernel<<<256, 256, 0, stream>>>(image, coords, bpack, partial);
    head_kernel<<<32, 256, 0, stream>>>(partial, keep, label_b, dir_b,
                                        out, score, label, validf, dirv);
    topk_kernel<<<1, 1024, 0, stream>>>(score, label, validf, dirv, coords, taken, out);
}
